// Causal_Atten_26637387170134
// MI455X (gfx1250) — compile-verified, matrix operand order checked by reading, not run on hardware
//
#include <hip/hip_runtime.h>

// ---------------------------------------------------------------------------
// Fused causal MHA forward for MI455X (gfx1250, wave32, WMMA).
//   1) cvt_f32_f16: x + four weights -> f16 workspace
//   2) gemm_xwT<0>: Q = (x @ Wq^T + bq) * sqrt(DH) -> f16 (B,H,N,DH)
//      gemm_xwT<0>: K                              -> f16 (B,H,N,DH)
//      gemm_xwT<2>: V                              -> f16 (B,H,DH,N) transposed
//   3) attn: flash attention, one wave per (b,h,16-query tile).
//      S^T = K Q^T and O^T = V^T P^T: the WMMA C layout of S^T is exactly the
//      B-fragment layout of the PV matmul (no LDS / no shuffles); all
//      fragment loads are forced to global_load_b128.
//   4) gemm_xwT<1>: out = Ctx @ Wp^T + bp, f32 to d_out
// ---------------------------------------------------------------------------

typedef __attribute__((ext_vector_type(16))) _Float16 v16h;
typedef __attribute__((ext_vector_type(8)))  _Float16 v8h;   // one b128 load
typedef __attribute__((ext_vector_type(8)))  float    v8f;

constexpr int Bb = 4, Ns = 2048, Dd = 1024, Hh = 16, DHd = 64;
constexpr int Mrows = Bb * Ns;           // 8192
constexpr float NEG_INF = -__builtin_inff();

__device__ __forceinline__ v8f wmma_f16(v16h a, v16h b, v8f c) {
  return __builtin_amdgcn_wmma_f32_16x16x32_f16(
      /*neg_a=*/false, a, /*neg_b=*/false, b,
      /*c_mod=*/(short)0, c, /*reuse_a=*/false, /*reuse_b=*/false);
}

// A/B fragment (16x32 f16, rows striped over lanes) from a row-major matrix.
// rowp + kbase must be 16-byte aligned (true at every call site); the two
// 8-half groups load as global_load_b128.
__device__ __forceinline__ v16h ldfrag_row(const _Float16* __restrict__ rowp,
                                           int kbase) {
  const v8h lo = *reinterpret_cast<const v8h*>(rowp + kbase);
  const v8h hi = *reinterpret_cast<const v8h*>(rowp + kbase + 16);
  v16h f;
#pragma unroll
  for (int j = 0; j < 8; ++j) { f[j] = lo[j]; f[j + 8] = hi[j]; }
  return f;
}

__global__ void cvt_f32_f16(const float* __restrict__ src,
                            _Float16* __restrict__ dst, int n) {
  int i = blockIdx.x * blockDim.x + threadIdx.x;
  if (i < n) dst[i] = (_Float16)src[i];
}

// Y = (X(8192xK) @ W^T(KxN) + bias) * oscale.  One wave per 64x32 block
// (4x2 WMMA tiles: 6 fragment loads feed 8 WMMAs per 32-deep k-step).
// MODE 0: f16 into (B,H,N,DH)   (Q with oscale=8, K with oscale=1)
// MODE 1: f32 into (Mrows x Dd) (final output)
// MODE 2: f16 into (B,H,DH,N)   (V, pre-transposed for the PV matmul)
template <int MODE>
__global__ __launch_bounds__(128) void gemm_xwT(
    const _Float16* __restrict__ X, const _Float16* __restrict__ W,
    const float* __restrict__ bias, float oscale,
    _Float16* __restrict__ outH, float* __restrict__ outF) {
  const int lane = threadIdx.x & 31;
  const int half = lane >> 4;
  const int ln = lane & 15;
  const int wave = blockIdx.x * (blockDim.x >> 5) + (threadIdx.x >> 5);
  constexpr int tilesN = Dd / 32;                    // 32
  const int m0 = (wave / tilesN) * 64;
  const int n0 = (wave % tilesN) * 32;

  const _Float16* xr[4];
#pragma unroll
  for (int i = 0; i < 4; ++i) xr[i] = X + (size_t)(m0 + 16 * i + ln) * Dd;
  const _Float16* wr0 = W + (size_t)(n0 + ln) * Dd;  // W row n == col n of W^T
  const _Float16* wr1 = W + (size_t)(n0 + 16 + ln) * Dd;

  v8f c[4][2];
#pragma unroll
  for (int i = 0; i < 4; ++i) { c[i][0] = (v8f){}; c[i][1] = (v8f){}; }

#pragma unroll 2
  for (int k0 = 0; k0 < Dd; k0 += 32) {
    const int kb = k0 + 8 * half;
    v16h b0 = ldfrag_row(wr0, kb);
    v16h b1 = ldfrag_row(wr1, kb);
#pragma unroll
    for (int i = 0; i < 4; ++i) {
      v16h a = ldfrag_row(xr[i], kb);
      c[i][0] = wmma_f16(a, b0, c[i][0]);
      c[i][1] = wmma_f16(a, b1, c[i][1]);
    }
  }

#pragma unroll
  for (int i = 0; i < 4; ++i) {
#pragma unroll
    for (int j = 0; j < 2; ++j) {
#pragma unroll
      for (int r = 0; r < 8; ++r) {
        const int m = m0 + 16 * i + r + 8 * half;  // C layout: reg+half -> row
        const int n = n0 + 16 * j + ln;            // lane%16 -> col
        const float v = (c[i][j][r] + bias[n]) * oscale;
        if (MODE == 1) {
          outF[(size_t)m * Dd + n] = v;
        } else {
          const int b = m >> 11, seq = m & (Ns - 1);
          const int h = n >> 6, dh = n & (DHd - 1);
          const size_t head = (size_t)(b * Hh + h);
          if (MODE == 0)
            outH[(head * Ns + seq) * DHd + dh] = (_Float16)v;
          else  // MODE 2: transposed per-head (DH x Ns)
            outH[(head * DHd + dh) * Ns + seq] = (_Float16)v;
        }
      }
    }
  }
}

// Flash attention, one wave per (b, h, 16-query tile).
// Vt is (B,H,DH,N): V^T rows are contiguous -> b128 fragment loads.
// The sqrt(DH) logit scale is pre-folded into Q.
__global__ __launch_bounds__(256) void attn_kernel(
    const _Float16* __restrict__ Q, const _Float16* __restrict__ K,
    const _Float16* __restrict__ Vt, _Float16* __restrict__ Ctx) {
  const int lane = threadIdx.x & 31;
  const int half = lane >> 4;
  const int ln = lane & 15;
  const int wave = blockIdx.x * (blockDim.x >> 5) + (threadIdx.x >> 5);
  constexpr int qTiles = Ns / 16;                 // 128
  const int bh = wave / qTiles;
  const int qblk = wave % qTiles;
  const int b = bh / Hh, h = bh % Hh;

  const _Float16* Qh = Q + (size_t)bh * Ns * DHd;
  const _Float16* Kh = K + (size_t)bh * Ns * DHd;
  const _Float16* Vh = Vt + (size_t)bh * DHd * Ns;

  const int qg = qblk * 16 + ln;                  // this lane's query column

  // B fragments of Q^T (dh x 16 queries): Q rows are contiguous.
  const _Float16* qrow = Qh + (size_t)qg * DHd;
  v16h bq0 = ldfrag_row(qrow, 8 * half);          // dh 0..31
  v16h bq1 = ldfrag_row(qrow, 32 + 8 * half);     // dh 32..63

  // V^T row pointers for the four 16-dh A tiles (rows striped over lanes)
  const _Float16* vr[4];
#pragma unroll
  for (int t = 0; t < 4; ++t) vr[t] = Vh + (size_t)(16 * t + ln) * Ns;

  v8f ot[4];
#pragma unroll
  for (int t = 0; t < 4; ++t) ot[t] = (v8f){};
  float m_run = NEG_INF, l_run = 0.f;

  const int jEnd = (qblk + 1) * 16;
  for (int jb = 0; jb < jEnd; jb += 32) {
    // prefetch next K block behind the softmax VALU work
    if (jb + 32 < jEnd)
      __builtin_prefetch(Kh + (size_t)(jb + 32 + ln) * DHd, 0, 1);

    const int kb0 = 8 * half, kb1 = 32 + 8 * half;
    const _Float16* kr0 = Kh + (size_t)(jb + ln) * DHd;
    const _Float16* kr1 = Kh + (size_t)(jb + 16 + ln) * DHd;
    v16h ak0l = ldfrag_row(kr0, kb0);
    v16h ak0h = ldfrag_row(kr0, kb1);
    v16h ak1l = ldfrag_row(kr1, kb0);
    v16h ak1h = ldfrag_row(kr1, kb1);

    // S^T = K Q^T, two 16x16 f32 tiles (keys jb..+15, jb+16..+31)
    v8f st0 = {}, st1 = {};
    st0 = wmma_f16(ak0l, bq0, st0);
    st0 = wmma_f16(ak0h, bq1, st0);
    st1 = wmma_f16(ak1l, bq0, st1);
    st1 = wmma_f16(ak1h, bq1, st1);

    // causal mask + per-query block max (lane = query, regs = keys)
    float p0[8], p1[8];
    float bmax = NEG_INF;
#pragma unroll
    for (int r = 0; r < 8; ++r) {
      const int key0 = jb + r + 8 * half;
      const int key1 = key0 + 16;
      const float s0 = (key0 <= qg) ? st0[r] : NEG_INF;
      const float s1 = (key1 <= qg) ? st1[r] : NEG_INF;
      p0[r] = s0; p1[r] = s1;
      bmax = fmaxf(bmax, fmaxf(s0, s1));
    }
    bmax = fmaxf(bmax, __shfl_xor(bmax, 16, 32));  // other key-half, same query

    const float m_new = fmaxf(m_run, bmax);
    const float sc = __expf(m_run - m_new);
    m_run = m_new;

    // P^T lands in exactly the B-fragment layout for the O^T WMMA
    float lsum = 0.f;
    v16h bp;
#pragma unroll
    for (int r = 0; r < 8; ++r) {
      const float e0 = __expf(p0[r] - m_new);
      const float e1 = __expf(p1[r] - m_new);
      lsum += e0 + e1;
      bp[r]     = (_Float16)e0;
      bp[r + 8] = (_Float16)e1;
    }
    lsum += __shfl_xor(lsum, 16, 32);
    l_run = l_run * sc + lsum;

#pragma unroll
    for (int t = 0; t < 4; ++t)
#pragma unroll
      for (int r = 0; r < 8; ++r) ot[t][r] *= sc;

    // O^T += V^T P^T  (V^T fragments are contiguous b128 loads)
    const int keyb = jb + 8 * half;
#pragma unroll
    for (int t = 0; t < 4; ++t) {
      v16h av = ldfrag_row(vr[t], keyb);
      ot[t] = wmma_f16(av, bp, ot[t]);
    }
  }

  // normalize and scatter into (B*Ns x Dd) context, f16
  const float inv = 1.0f / l_run;
  _Float16* crow = Ctx + ((size_t)b * Ns + qblk * 16 + ln) * Dd + h * DHd;
#pragma unroll
  for (int t = 0; t < 4; ++t)
#pragma unroll
    for (int r = 0; r < 8; ++r)
      crow[16 * t + r + 8 * half] = (_Float16)(ot[t][r] * inv);
}

extern "C" void kernel_launch(void* const* d_in, const int* in_sizes, int n_in,
                              void* d_out, int out_size, void* d_ws,
                              size_t ws_size, hipStream_t stream) {
  const float* x  = (const float*)d_in[0];
  const float* Wq = (const float*)d_in[1];
  const float* bq = (const float*)d_in[2];
  const float* Wk = (const float*)d_in[3];
  const float* bk = (const float*)d_in[4];
  const float* Wv = (const float*)d_in[5];
  const float* bv = (const float*)d_in[6];
  const float* Wp = (const float*)d_in[7];
  const float* bp = (const float*)d_in[8];
  float* out = (float*)d_out;

  const size_t nX = (size_t)Mrows * Dd;   // 8.39M
  const size_t nW = (size_t)Dd * Dd;      // 1.05M
  _Float16* Xh  = (_Float16*)d_ws;
  _Float16* Wqh = Xh + nX;
  _Float16* Wkh = Wqh + nW;
  _Float16* Wvh = Wkh + nW;
  _Float16* Wph = Wvh + nW;
  _Float16* Qh  = Wph + nW;   // (B,H,N,DH), pre-scaled by sqrt(DH)
  _Float16* Kh  = Qh + nX;    // (B,H,N,DH)
  _Float16* Vth = Kh + nX;    // (B,H,DH,N)
  _Float16* Ctx = Vth + nX;   // (Mrows x Dd)

  // 1) f32 -> f16 conversions
  cvt_f32_f16<<<(int)((nX + 255) / 256), 256, 0, stream>>>(x, Xh, (int)nX);
  cvt_f32_f16<<<(int)((nW + 255) / 256), 256, 0, stream>>>(Wq, Wqh, (int)nW);
  cvt_f32_f16<<<(int)((nW + 255) / 256), 256, 0, stream>>>(Wk, Wkh, (int)nW);
  cvt_f32_f16<<<(int)((nW + 255) / 256), 256, 0, stream>>>(Wv, Wvh, (int)nW);
  cvt_f32_f16<<<(int)((nW + 255) / 256), 256, 0, stream>>>(Wp, Wph, (int)nW);

  // 2) QKV projections: (8192/64)*(1024/32) = 4096 waves, 4 waves/block
  const int gemmBlocks = (Mrows / 64) * (Dd / 32) / 4;  // 1024
  gemm_xwT<0><<<gemmBlocks, 128, 0, stream>>>(Xh, Wqh, bq, 8.0f, Qh, nullptr);
  gemm_xwT<0><<<gemmBlocks, 128, 0, stream>>>(Xh, Wkh, bk, 1.0f, Kh, nullptr);
  gemm_xwT<2><<<gemmBlocks, 128, 0, stream>>>(Xh, Wvh, bv, 1.0f, Vth, nullptr);

  // 3) attention: B*H * (N/16) = 8192 waves, 8 waves/block
  const int attnBlocks = Bb * Hh * (Ns / 16) / 8;       // 1024
  attn_kernel<<<attnBlocks, 256, 0, stream>>>(Qh, Kh, Vth, Ctx);

  // 4) output projection, f32
  gemm_xwT<1><<<gemmBlocks, 128, 0, stream>>>(Ctx, Wph, bp, 1.0f, nullptr, out);
}
